// RWKV_Tmix_x070_64089501991163
// MI455X (gfx1250) — matrix operand order repaired and checked by reading, not compiled, not run on hardware
//
#include <hip/hip_runtime.h>
#include <hip/hip_bf16.h>
#include <math.h>

// ---------------------------------------------------------------------------
// RWKV-7 Tmix forward for MI455X (gfx1250, wave32, WMMA + async-to-LDS).
// ---------------------------------------------------------------------------

#define Bb 4
#define Tt 2048
#define Cc 1024
#define Hh 16
#define Nn 64
#define Mm (Bb * Tt)          // 8192 token rows
#define BTC ((size_t)Mm * Cc) // 8388608
#define EPS_LN 6.4e-4f        // 1e-5 * 8^2

typedef __attribute__((ext_vector_type(16))) __bf16 v16bf;
typedef __attribute__((ext_vector_type(8)))  __bf16 v8bf;
typedef __attribute__((ext_vector_type(8)))  float  v8f;

__device__ __forceinline__ float sigmoidf_(float x) { return 1.0f / (1.0f + __expf(-x)); }

#ifndef __has_builtin
#define __has_builtin(x) 0
#endif

#if __has_builtin(__builtin_amdgcn_global_load_async_to_lds_b128)
#define ASYNC_CP 1
#else
#define ASYNC_CP 0
#endif

typedef int v4i_ __attribute__((vector_size(16)));
typedef __attribute__((address_space(1))) v4i_* as1_v4i;
typedef __attribute__((address_space(3))) v4i_* as3_v4i;

__device__ __forceinline__ void async_cp16(const __bf16* g, __bf16* l) {
#if ASYNC_CP
  __builtin_amdgcn_global_load_async_to_lds_b128(
      (as1_v4i)(__UINTPTR_TYPE__)g,
      (as3_v4i)(unsigned)(__UINTPTR_TYPE__)l,
      0, 0);
#else
  *(uint4*)l = *(const uint4*)g;
#endif
}

__device__ __forceinline__ void wait_async_all() {
#if ASYNC_CP
#if __has_builtin(__builtin_amdgcn_s_wait_asynccnt)
  __builtin_amdgcn_s_wait_asynccnt(0);
#else
  asm volatile("s_wait_asynccnt 0x0" ::: "memory");
#endif
#endif
}

// ---------------------------------------------------------------------------
// Converts / activations
// ---------------------------------------------------------------------------
__global__ void k_f32_to_bf16(const float* __restrict__ src, __bf16* __restrict__ dst, int n) {
  int i = blockIdx.x * blockDim.x + threadIdx.x;
  if (i < n) dst[i] = (__bf16)src[i];
}

// dst[cols][drows] = transpose(src[rows][cols]), zero-padded rows >= rows.
__global__ void k_f32_to_bf16_T(const float* __restrict__ src, __bf16* __restrict__ dst,
                                int rows, int cols, int drows) {
  int i = blockIdx.x * blockDim.x + threadIdx.x;
  if (i < cols * drows) {
    int c = i / drows, r = i % drows;
    dst[i] = (r < rows) ? (__bf16)src[(size_t)r * cols + c] : (__bf16)0.0f;
  }
}

// dst[M][dcols] = act(src[M][cols]) with zero pad for col >= cols.
// mode: 0 = copy, 1 = tanh, 2 = sigmoid
__global__ void k_act_to_bf16(const float* __restrict__ src, __bf16* __restrict__ dst,
                              int m, int cols, int dcols, int mode) {
  int i = blockIdx.x * blockDim.x + threadIdx.x;
  if (i < m * dcols) {
    int row = i / dcols, col = i % dcols;
    float v = 0.0f;
    if (col < cols) {
      v = src[(size_t)row * cols + col];
      if (mode == 1) v = tanhf(v);
      else if (mode == 2) v = sigmoidf_(v);
    }
    dst[i] = (__bf16)v;
  }
}

// ---------------------------------------------------------------------------
// Token-shift mixing: one block per token, 256 threads, 4 channels/thread.
// ---------------------------------------------------------------------------
__global__ void k_prep_mix(const float* __restrict__ x, const float* __restrict__ mask,
                           const float* __restrict__ cr, const float* __restrict__ cw,
                           const float* __restrict__ ck, const float* __restrict__ cv,
                           const float* __restrict__ ca, const float* __restrict__ cg,
                           __bf16* __restrict__ xr, __bf16* __restrict__ xw,
                           __bf16* __restrict__ xk, __bf16* __restrict__ xv,
                           __bf16* __restrict__ xa, __bf16* __restrict__ xg) {
  int tok = blockIdx.x;
  int t   = tok % Tt;
  float m  = mask[tok];
  float mp = (t > 0) ? mask[tok - 1] : 0.0f;
  size_t base = (size_t)tok * Cc;
  #pragma unroll
  for (int u = 0; u < 4; u++) {
    int c = threadIdx.x * 4 + u;
    float xc = x[base + c] * m;
    float xp = (t > 0) ? x[base - Cc + c] * mp : 0.0f;
    float xx = xp - xc;
    xr[base + c] = (__bf16)(xc + xx * cr[c]);
    xw[base + c] = (__bf16)(xc + xx * cw[c]);
    xk[base + c] = (__bf16)(xc + xx * ck[c]);
    xv[base + c] = (__bf16)(xc + xx * cv[c]);
    xa[base + c] = (__bf16)(xc + xx * ca[c]);
    xg[base + c] = (__bf16)(xc + xx * cg[c]);
  }
}

// ---------------------------------------------------------------------------
// Big WMMA GEMM (N multiple of 128, K multiple of 64).
// Out[M,N] = A[M,K](bf16) * Bt[N,K](bf16)^T, fp32 accum.
// Block 256 thr = 8 waves; block tile 128x128; wave tile 32x64.
// A/B tiles (128 x 64 bf16) staged via async-to-LDS, double buffered:
// 16 WMMAs per wait_asynccnt+barrier. LDS row stride 80 bf16 (160 B) keeps
// v16bf 32B-aligned and spreads banks. 80 KB LDS -> 4 blocks/WGP (320 KB pool).
// ---------------------------------------------------------------------------
#define KSTEP 64
#define LSTR  80

__global__ void k_gemm_big(const __bf16* __restrict__ A, const __bf16* __restrict__ Bt,
                           float* __restrict__ Out, int N, int K) {
  __shared__ __bf16 lA[2][128][LSTR];
  __shared__ __bf16 lB[2][128][LSTR];

  int tid  = threadIdx.x;
  int lane = tid & 31, wave = tid >> 5;
  int half = lane >> 4, l16 = lane & 15;
  int wm = wave >> 1, wn = wave & 1;     // wave tile: rows wm*32, cols wn*64
  int m0b = blockIdx.x * 128, n0b = blockIdx.y * 128;

  v8f acc[8] = {};   // [mt*4 + nt]

  // Loader: tile = 128 rows x 64 bf16 (128 B) = 1024 x 16B chunks;
  // 4 chunks per thread for A and 4 for B.
  auto issue = [&](int buf, int ks) {
    #pragma unroll
    for (int j = 0; j < 4; j++) {
      int q = tid + 256 * j;
      int row = q >> 3, cq = q & 7;
      async_cp16(A  + (size_t)(m0b + row) * K + ks * KSTEP + cq * 8, &lA[buf][row][cq * 8]);
      async_cp16(Bt + (size_t)(n0b + row) * K + ks * KSTEP + cq * 8, &lB[buf][row][cq * 8]);
    }
  };

  int nks = K >> 6;
  issue(0, 0);
  for (int ks = 0; ks < nks; ks++) {
    int buf = ks & 1;
    wait_async_all();
    __syncthreads();
    if (ks + 1 < nks) issue(buf ^ 1, ks + 1);

    #pragma unroll
    for (int kk = 0; kk < KSTEP; kk += 32) {
      // A fragments: rows (wm*32 + mt*16 + l16); K = kk+half*16..+15
      v16bf af[2];
      #pragma unroll
      for (int mt = 0; mt < 2; mt++) {
        const __bf16* ar = &lA[buf][wm * 32 + mt * 16 + l16][kk + half * 16];
        v8bf lo = *(const v8bf*)ar;
        v8bf hi = *(const v8bf*)(ar + 8);
        #pragma unroll
        for (int i = 0; i < 8; i++) { af[mt][i] = lo[i]; af[mt][i + 8] = hi[i]; }
      }
      // Preload all 4 B fragments, then run the 8-WMMA chain back-to-back.
      v16bf bfr[4];
      #pragma unroll
      for (int nt = 0; nt < 4; nt++)
        bfr[nt] = *(const v16bf*)&lB[buf][wn * 64 + nt * 16 + l16][kk + half * 16];
      #pragma unroll
      for (int nt = 0; nt < 4; nt++) {
        acc[0 * 4 + nt] = __builtin_amdgcn_wmma_f32_16x16x32_bf16(
            false, af[0], false, bfr[nt], (short)0, acc[0 * 4 + nt], false, false);
        acc[1 * 4 + nt] = __builtin_amdgcn_wmma_f32_16x16x32_bf16(
            false, af[1], false, bfr[nt], (short)0, acc[1 * 4 + nt], false, false);
      }
    }
  }

  #pragma unroll
  for (int mt = 0; mt < 2; mt++) {
    #pragma unroll
    for (int nt = 0; nt < 4; nt++) {
      int n = n0b + wn * 64 + nt * 16 + l16;
      #pragma unroll
      for (int vg = 0; vg < 8; vg++) {
        int m = m0b + wm * 32 + mt * 16 + half * 8 + vg;
        Out[(size_t)m * N + n] = acc[mt * 4 + nt][vg];
      }
    }
  }
}

// ---------------------------------------------------------------------------
// Small-N WMMA GEMM (LoRA stage-1). NT tiles of 16 cols per wave, branch-free.
// Wave = 16 rows; block = 8 waves = 128 rows.
// ---------------------------------------------------------------------------
template <int NT>
__global__ void k_gemm_small(const __bf16* __restrict__ A, const __bf16* __restrict__ Bt,
                             float* __restrict__ Out, int N, int K, int n_base) {
  int lane = threadIdx.x & 31, wave = threadIdx.x >> 5;
  int half = lane >> 4, l16 = lane & 15;
  int m0 = blockIdx.x * 128 + wave * 16;
  int n0 = n_base + blockIdx.y * 64;

  v8f acc[NT] = {};
  const __bf16* Arow = A + (size_t)(m0 + l16) * K;
  const __bf16* Bcol[NT];
  #pragma unroll
  for (int t = 0; t < NT; t++) Bcol[t] = Bt + (size_t)(n0 + t * 16 + l16) * K;

  for (int k0 = 0; k0 < K; k0 += 32) {
    v8bf lo = *(const v8bf*)(Arow + k0 + half * 16);
    v8bf hi = *(const v8bf*)(Arow + k0 + half * 16 + 8);
    v16bf afrag;
    #pragma unroll
    for (int i = 0; i < 8; i++) { afrag[i] = lo[i]; afrag[i + 8] = hi[i]; }
    #pragma unroll
    for (int t = 0; t < NT; t++) {
      v16bf bfrag = *(const v16bf*)(Bcol[t] + k0 + half * 16);
      acc[t] = __builtin_amdgcn_wmma_f32_16x16x32_bf16(
          false, afrag, false, bfrag, (short)0, acc[t], false, false);
    }
  }
  #pragma unroll
  for (int t = 0; t < NT; t++) {
    int n = n0 + t * 16 + l16;
    #pragma unroll
    for (int vg = 0; vg < 8; vg++) {
      int m = m0 + half * 8 + vg;
      Out[(size_t)m * N + n] = acc[t][vg];
    }
  }
}

// ---------------------------------------------------------------------------
// Post-projection fusion (kk-norm via 16-lane shuffles, k/a/v/w finalize).
// ---------------------------------------------------------------------------
__global__ void k_post(float* __restrict__ k, float* __restrict__ v,
                       float* __restrict__ wtmp, float* __restrict__ atmp,
                       const float* __restrict__ vtmp, float* __restrict__ kk,
                       const float* __restrict__ vfirst, const float* __restrict__ mask,
                       const float* __restrict__ k_k, const float* __restrict__ k_a,
                       const float* __restrict__ w0, const float* __restrict__ a0,
                       const float* __restrict__ v0) {
  int tok = blockIdx.x;
  size_t base = (size_t)tok * Cc;
  float mval = mask[tok];
  int c0 = threadIdx.x * 4;

  float kkv[4], kv[4];
  float ss = 0.0f;
  #pragma unroll
  for (int u = 0; u < 4; u++) {
    int c = c0 + u;
    kv[u]  = k[base + c];
    kkv[u] = kv[u] * k_k[c];
    ss += kkv[u] * kkv[u];
  }
  ss += __shfl_xor(ss, 1); ss += __shfl_xor(ss, 2);
  ss += __shfl_xor(ss, 4); ss += __shfl_xor(ss, 8);
  float scale = 1.0f / fmaxf(sqrtf(ss), 1e-12f);

  #pragma unroll
  for (int u = 0; u < 4; u++) {
    int c = c0 + u;
    size_t i = base + c;
    kk[i] = kkv[u] * scale;
    float av = sigmoidf_(a0[c] + atmp[i]);
    atmp[i] = av;
    k[i] = kv[u] * (1.0f + (av - 1.0f) * k_a[c]);
    float vfac = sigmoidf_(v0[c] + vtmp[i]);
    float vv = v[i];
    v[i] = (vv + (vfirst[i] - vv) * vfac) * mval;
    wtmp[i] = w0[c] + wtmp[i];
  }
}

// ---------------------------------------------------------------------------
// wkv7 recurrence: block per (b,h); state in VGPRs; LDS-staged step vectors.
// ---------------------------------------------------------------------------
__global__ void k_wkv7(const float* __restrict__ r, const float* __restrict__ w,
                       const float* __restrict__ k, const float* __restrict__ v,
                       const float* __restrict__ a_sig, const float* __restrict__ kk,
                       float* __restrict__ y) {
  int b = blockIdx.x / Hh;
  int h = blockIdx.x % Hh;
  int tid = threadIdx.x;
  int i  = tid >> 2;
  int jb = (tid & 3) * 16;

  float S[16];
  #pragma unroll
  for (int jj = 0; jj < 16; jj++) S[jj] = 0.0f;

  __shared__ float shr[Nn], shw[Nn], shk[Nn], shv[Nn], sha[Nn], shb[Nn];

  for (int t = 0; t < Tt; t++) {
    size_t base = ((size_t)(b * Tt + t)) * Cc + h * Nn;
    if (tid < Nn) {
      int j = tid;
      float kkj = kk[base + j];
      shr[j] = r[base + j];
      shw[j] = __expf(-__expf(w[base + j]));
      shk[j] = k[base + j];
      shv[j] = v[base + j];
      sha[j] = -kkj;
      shb[j] = kkj * a_sig[base + j];
    }
    __syncthreads();

    float sa = 0.0f;
    #pragma unroll
    for (int jj = 0; jj < 16; jj++) sa += S[jj] * sha[jb + jj];
    sa += __shfl_xor(sa, 1);
    sa += __shfl_xor(sa, 2);

    float vi = shv[i];
    float yi = 0.0f;
    #pragma unroll
    for (int jj = 0; jj < 16; jj++) {
      int j = jb + jj;
      float s = S[jj] * shw[j] + sa * shb[j] + vi * shk[j];
      S[jj] = s;
      yi += s * shr[j];
    }
    yi += __shfl_xor(yi, 1);
    yi += __shfl_xor(yi, 2);
    if ((tid & 3) == 0) y[base + i] = yi;
    __syncthreads();
  }
}

// ---------------------------------------------------------------------------
// GroupNorm(H) + rk bonus + gate -> bf16 "gated".
// ---------------------------------------------------------------------------
__global__ void k_norm_gate(const float* __restrict__ y, const float* __restrict__ r,
                            const float* __restrict__ k, const float* __restrict__ v,
                            const float* __restrict__ g, const float* __restrict__ ln_g,
                            const float* __restrict__ ln_b, const float* __restrict__ r_k,
                            __bf16* __restrict__ gated) {
  int tok = blockIdx.x;
  size_t base = (size_t)tok * Cc;
  int c0 = threadIdx.x * 4;

  float yv[4];
  float sum = 0.0f, sumsq = 0.0f, rk = 0.0f;
  #pragma unroll
  for (int u = 0; u < 4; u++) {
    int c = c0 + u;
    size_t idx = base + c;
    yv[u] = y[idx];
    sum += yv[u];
    sumsq += yv[u] * yv[u];
    rk += r[idx] * k[idx] * r_k[c];
  }
  sum   += __shfl_xor(sum, 1);   sum   += __shfl_xor(sum, 2);
  sum   += __shfl_xor(sum, 4);   sum   += __shfl_xor(sum, 8);
  sumsq += __shfl_xor(sumsq, 1); sumsq += __shfl_xor(sumsq, 2);
  sumsq += __shfl_xor(sumsq, 4); sumsq += __shfl_xor(sumsq, 8);
  rk    += __shfl_xor(rk, 1);    rk    += __shfl_xor(rk, 2);
  rk    += __shfl_xor(rk, 4);    rk    += __shfl_xor(rk, 8);

  float mu  = sum * (1.0f / Nn);
  float var = sumsq * (1.0f / Nn) - mu * mu;
  float inv = rsqrtf(var + EPS_LN);

  #pragma unroll
  for (int u = 0; u < 4; u++) {
    int c = c0 + u;
    size_t idx = base + c;
    float yn = (yv[u] - mu) * inv * ln_g[c] + ln_b[c];
    yn += rk * v[idx];
    gated[idx] = (__bf16)(yn * g[idx]);
  }
}

// ---------------------------------------------------------------------------
// Host launch
// ---------------------------------------------------------------------------
extern "C" void kernel_launch(void* const* d_in, const int* in_sizes, int n_in,
                              void* d_out, int out_size, void* d_ws, size_t ws_size,
                              hipStream_t stream) {
  (void)in_sizes; (void)n_in; (void)out_size; (void)ws_size;
  const float* x     = (const float*)d_in[0];
  const float* mask  = (const float*)d_in[1];
  const float* vfst  = (const float*)d_in[2];
  const float* x_r   = (const float*)d_in[3];
  const float* x_w   = (const float*)d_in[4];
  const float* x_k   = (const float*)d_in[5];
  const float* x_v   = (const float*)d_in[6];
  const float* x_a   = (const float*)d_in[7];
  const float* x_g   = (const float*)d_in[8];
  const float* w0    = (const float*)d_in[9];
  const float* w1    = (const float*)d_in[10];
  const float* w2    = (const float*)d_in[11];
  const float* a0    = (const float*)d_in[12];
  const float* a1    = (const float*)d_in[13];
  const float* a2    = (const float*)d_in[14];
  const float* v0    = (const float*)d_in[15];
  const float* v1    = (const float*)d_in[16];
  const float* v2    = (const float*)d_in[17];
  const float* g1    = (const float*)d_in[18];
  const float* g2    = (const float*)d_in[19];
  const float* k_k   = (const float*)d_in[20];
  const float* k_a   = (const float*)d_in[21];
  const float* r_k   = (const float*)d_in[22];
  const float* W_r   = (const float*)d_in[23];
  const float* W_k   = (const float*)d_in[24];
  const float* W_v   = (const float*)d_in[25];
  const float* W_o   = (const float*)d_in[26];
  const float* ln_g  = (const float*)d_in[27];
  const float* ln_b  = (const float*)d_in[28];
  float* out = (float*)d_out;

  char* wp = (char*)d_ws;
  auto alloc = [&](size_t bytes) -> void* {
    void* p = (void*)wp;
    wp += (bytes + 255) & ~(size_t)255;
    return p;
  };

  // LoRA dims; stage-2 K padded to multiples of 64 for the staged GEMM.
  const int DW = 64, DA = 64, DV = 32, DG = 160;
  const int DWp = 64, DAp = 64, DVp = 64, DGp = 192;

  __bf16* xr = (__bf16*)alloc(BTC * 2);
  __bf16* xw = (__bf16*)alloc(BTC * 2);
  __bf16* xk = (__bf16*)alloc(BTC * 2);
  __bf16* xv = (__bf16*)alloc(BTC * 2);
  __bf16* xa = (__bf16*)alloc(BTC * 2);
  __bf16* xg = (__bf16*)alloc(BTC * 2);
  __bf16* Wr_b = (__bf16*)alloc((size_t)Cc * Cc * 2);
  __bf16* Wk_b = (__bf16*)alloc((size_t)Cc * Cc * 2);
  __bf16* Wv_b = (__bf16*)alloc((size_t)Cc * Cc * 2);
  __bf16* Wo_b = (__bf16*)alloc((size_t)Cc * Cc * 2);
  __bf16* w1T = (__bf16*)alloc((size_t)Cc * DW  * 2);  // [DW,  C]
  __bf16* w2T = (__bf16*)alloc((size_t)Cc * DWp * 2);  // [C, DWp]
  __bf16* a1T = (__bf16*)alloc((size_t)Cc * DA  * 2);
  __bf16* a2T = (__bf16*)alloc((size_t)Cc * DAp * 2);
  __bf16* v1T = (__bf16*)alloc((size_t)Cc * DV  * 2);
  __bf16* v2T = (__bf16*)alloc((size_t)Cc * DVp * 2);
  __bf16* g1T = (__bf16*)alloc((size_t)Cc * DG  * 2);
  __bf16* g2T = (__bf16*)alloc((size_t)Cc * DGp * 2);
  float* rbuf = (float*)alloc(BTC * 4);
  float* kbuf = (float*)alloc(BTC * 4);
  float* vbuf = (float*)alloc(BTC * 4);
  float* wtmp = (float*)alloc(BTC * 4);
  float* atmp = (float*)alloc(BTC * 4);
  float* vtmp = (float*)alloc(BTC * 4);
  float* kkb  = (float*)alloc(BTC * 4);
  float* gbuf = (float*)alloc(BTC * 4);
  float* ybuf = (float*)alloc(BTC * 4);
  float*  s1f = (float*)alloc((size_t)Mm * DG  * 4);
  __bf16* s1b = (__bf16*)alloc((size_t)Mm * DGp * 2);
  __bf16* gated = xr;  // xr free after the r projection

  const int TPB = 256;
  auto cdiv = [](int a, int b) { return (a + b - 1) / b; };
  auto big = [&](const __bf16* A, const __bf16* Bt, float* Out, int N, int K) {
    dim3 g(Mm / 128, N / 128);
    k_gemm_big<<<g, TPB, 0, stream>>>(A, Bt, Out, N, K);
  };

  // 1) weight conversion ([N,K] layouts for WMMA B fragments; padded stage-2 K)
  int nCC = Cc * Cc;
  k_f32_to_bf16<<<cdiv(nCC, TPB), TPB, 0, stream>>>(W_r, Wr_b, nCC);
  k_f32_to_bf16<<<cdiv(nCC, TPB), TPB, 0, stream>>>(W_k, Wk_b, nCC);
  k_f32_to_bf16<<<cdiv(nCC, TPB), TPB, 0, stream>>>(W_v, Wv_b, nCC);
  k_f32_to_bf16<<<cdiv(nCC, TPB), TPB, 0, stream>>>(W_o, Wo_b, nCC);
  k_f32_to_bf16_T<<<cdiv(DW * Cc, TPB), TPB, 0, stream>>>(w1, w1T, Cc, DW, Cc);   // [C,DW]->[DW][C]
  k_f32_to_bf16_T<<<cdiv(Cc * DWp, TPB), TPB, 0, stream>>>(w2, w2T, DW, Cc, DWp); // [DW,C]->[C][DWp]
  k_f32_to_bf16_T<<<cdiv(DA * Cc, TPB), TPB, 0, stream>>>(a1, a1T, Cc, DA, Cc);
  k_f32_to_bf16_T<<<cdiv(Cc * DAp, TPB), TPB, 0, stream>>>(a2, a2T, DA, Cc, DAp);
  k_f32_to_bf16_T<<<cdiv(DV * Cc, TPB), TPB, 0, stream>>>(v1, v1T, Cc, DV, Cc);
  k_f32_to_bf16_T<<<cdiv(Cc * DVp, TPB), TPB, 0, stream>>>(v2, v2T, DV, Cc, DVp);
  k_f32_to_bf16_T<<<cdiv(DG * Cc, TPB), TPB, 0, stream>>>(g1, g1T, Cc, DG, Cc);
  k_f32_to_bf16_T<<<cdiv(Cc * DGp, TPB), TPB, 0, stream>>>(g2, g2T, DG, Cc, DGp);

  // 2) token-shift mixing
  k_prep_mix<<<Mm, TPB, 0, stream>>>(x, mask, x_r, x_w, x_k, x_v, x_a, x_g,
                                     xr, xw, xk, xv, xa, xg);

  // 3) dense projections (staged WMMA)
  big(xr, Wr_b, rbuf, Cc, Cc);
  big(xk, Wk_b, kbuf, Cc, Cc);
  big(xv, Wv_b, vbuf, Cc, Cc);

  // 4) LoRA pairs: stage1 (small-N, branch-free) -> act (pad K) -> stage2 (staged)
  k_gemm_small<4><<<dim3(Mm / 128, 1), TPB, 0, stream>>>(xw, w1T, s1f, DW, Cc, 0);
  k_act_to_bf16<<<cdiv(Mm * DWp, TPB), TPB, 0, stream>>>(s1f, s1b, Mm, DW, DWp, 1);
  big(s1b, w2T, wtmp, Cc, DWp);

  k_gemm_small<4><<<dim3(Mm / 128, 1), TPB, 0, stream>>>(xa, a1T, s1f, DA, Cc, 0);
  k_act_to_bf16<<<cdiv(Mm * DAp, TPB), TPB, 0, stream>>>(s1f, s1b, Mm, DA, DAp, 0);
  big(s1b, a2T, atmp, Cc, DAp);

  k_gemm_small<2><<<dim3(Mm / 128, 1), TPB, 0, stream>>>(xv, v1T, s1f, DV, Cc, 0);
  k_act_to_bf16<<<cdiv(Mm * DVp, TPB), TPB, 0, stream>>>(s1f, s1b, Mm, DV, DVp, 0);
  big(s1b, v2T, vtmp, Cc, DVp);

  k_gemm_small<4><<<dim3(Mm / 128, 2), TPB, 0, stream>>>(xg, g1T, s1f, DG, Cc, 0);
  k_gemm_small<2><<<dim3(Mm / 128, 1), TPB, 0, stream>>>(xg, g1T, s1f, DG, Cc, 128);
  k_act_to_bf16<<<cdiv(Mm * DGp, TPB), TPB, 0, stream>>>(s1f, s1b, Mm, DG, DGp, 2);
  big(s1b, g2T, gbuf, Cc, DGp);

  // 5) fused post-projection
  k_post<<<Mm, TPB, 0, stream>>>(kbuf, vbuf, wtmp, atmp, vtmp, kkb,
                                 vfst, mask, k_k, k_a, w0, a0, v0);

  // 6) wkv7 recurrence
  k_wkv7<<<Bb * Hh, TPB, 0, stream>>>(rbuf, wtmp, kbuf, vbuf, atmp, kkb, ybuf);

  // 7) GroupNorm + rk bonus + gate
  k_norm_gate<<<Mm, TPB, 0, stream>>>(ybuf, rbuf, kbuf, vbuf, gbuf,
                                      ln_g, ln_b, r_k, gated);

  // 8) output projection into d_out
  big(gated, Wo_b, out, Cc, Cc);
}
